// SoundSTGCN_23433341567345
// MI455X (gfx1250) — matrix operand order repaired and checked by reading, not compiled, not run on hardware
//
#include <hip/hip_runtime.h>

// ---------------- CDNA5 WMMA types ----------------
typedef __attribute__((ext_vector_type(16))) __bf16 v16bf;
typedef __attribute__((ext_vector_type(8)))  __bf16 bf16x8;
typedef __attribute__((ext_vector_type(8)))  float  v8f;

#define BATCH 32
#define NNODE 40
#define TLEN  174
#define BN    (BATCH * NNODE)        // 1280
#define RROWS (BN * TLEN)            // 222720 = 13920 * 16

__device__ __forceinline__ v8f wmma_bf16(v16bf a, v16bf b, v8f c) {
  return __builtin_amdgcn_wmma_f32_16x16x32_bf16(
      false, a, false, b, (short)0, c, false, false);
}

// two contiguous 8-elem (16B) loads fill one 16-bit fragment
__device__ __forceinline__ void frag_load8(v16bf& f, int off, const __bf16* p) {
  bf16x8 v = *(const bf16x8*)p;
#pragma unroll
  for (int e = 0; e < 8; ++e) f[off + e] = v[e];
}

// A-fragment K index (scalar fallback path), ISA 7.12.2 16-bit A 16x32
__device__ __forceinline__ int a_kmap(int e, int half) {
  int v = e >> 1, p = e & 1;
  int k = (v < 4) ? (2 * v + p) : (16 + 2 * (v - 4) + p);
  return k + 8 * half;
}

// ---------------- fp32 -> bf16 cast ----------------
__global__ void cast_f32_bf16(const float* __restrict__ src, __bf16* __restrict__ dst, int n) {
  int i = blockIdx.x * blockDim.x + threadIdx.x;
  if (i < n) dst[i] = (__bf16)src[i];
}

// ---------------- weight packing ----------------
// tconv: w[Cout,Cin,3] -> Wp[kw][Cout][Cin]  (K contiguous per row)
__global__ void pack_tconv_w(const float* __restrict__ w, __bf16* __restrict__ Wp,
                             int Cin, int Cout) {
  int o = blockIdx.x * 256 + threadIdx.x;
  int n = 3 * Cout * Cin;
  if (o >= n) return;
  int kw = o / (Cout * Cin);
  int rem = o - kw * Cout * Cin;
  int co = rem / Cin, ci = rem - co * Cin;
  Wp[o] = (__bf16)w[(co * Cin + ci) * 3 + kw];
}

// cheb mix: w[3C, C] -> Wpk[(q*C+co)*32 + j] = w[q*32+j][co]  (B-fragment contiguous)
__global__ void pack_mix_w(const float* __restrict__ w, __bf16* __restrict__ Wpk, int C) {
  int o = blockIdx.x * 256 + threadIdx.x;
  int n = 3 * C * C;
  if (o >= n) return;
  int q = o / (C * 32);
  int rem = o - q * C * 32;
  int co = rem >> 5, j = rem & 31;
  Wpk[o] = (__bf16)w[(q * 32 + j) * C + co];
}

// ---------------- per-bn [C,T] -> [T,C] transpose (LDS tiled) ----------------
__global__ __launch_bounds__(256)
void transpose_ct(const __bf16* __restrict__ in, __bf16* __restrict__ out, int C) {
  __shared__ __bf16 tile[16][17];
  const int bn = blockIdx.x, c0 = blockIdx.y << 4, t0 = blockIdx.z << 4;
  const int tx = threadIdx.x & 15, ty = threadIdx.x >> 4;
  const __bf16* ib = in + (size_t)bn * C * TLEN;
  int t = t0 + tx;
  tile[ty][tx] = (t < TLEN) ? ib[(size_t)(c0 + ty) * TLEN + t] : (__bf16)0.0f;
  __syncthreads();
  int tw = t0 + ty;
  if (tw < TLEN) out[(size_t)bn * TLEN * C + (size_t)tw * C + (c0 + tx)] = tile[tx][ty];
}

// ---------------- scalar tconv (block1 only: Cin=1, negligible FLOPs) ----------------
__global__ __launch_bounds__(32)
void tconv_wmma_scalar(const __bf16* __restrict__ X, const __bf16* __restrict__ W,
                       const float* __restrict__ bias, __bf16* __restrict__ Y,
                       int Cin, int Cout) {
  const int bn = blockIdx.x, co0 = blockIdx.y << 4, t0 = blockIdx.z << 4;
  const int lane = threadIdx.x & 31, half = lane >> 4, mn = lane & 15;
  const int K = Cin * 3;
  const __bf16* Xb = X + (size_t)bn * Cin * TLEN;
  const __bf16 zero = (__bf16)0.0f;
  v8f acc = {};
  for (int kk = 0; kk < K; kk += 32) {
    v16bf a, b;
    const int row = co0 + mn;
#pragma unroll
    for (int e = 0; e < 16; ++e) {
      int k = a_kmap(e, half) + kk;
      a[e] = (k < K) ? W[(size_t)row * K + k] : zero;
    }
#pragma unroll
    for (int e = 0; e < 16; ++e) {
      int k = a_kmap(e, half) + kk;
      __bf16 val = zero;
      if (k < K) {
        int ci = k / 3, kw = k - ci * 3;
        int tt = t0 + mn + kw - 1;
        if (tt >= 0 && tt < TLEN) val = Xb[(size_t)ci * TLEN + tt];
      }
      b[e] = val;
    }
    acc = wmma_bf16(a, b, acc);
  }
  const int t = t0 + mn;
  if (t < TLEN) {
#pragma unroll
    for (int j = 0; j < 8; ++j) {
      int co = co0 + j + 8 * half;
      if (co < Cout)
        Y[((size_t)bn * Cout + co) * TLEN + t] = (__bf16)(acc[j] + bias[co]);
    }
  }
}

// ---------------- vector tconv: split into 3 shifted GEMMs over K=Cin ----------------
// Xt [BN,T,Cin] (channel-contiguous), Wp [kw][Cout][Cin], Y flat [BN,Cout,T].
// One wave computes a 64(co) x 16(t) tile with 4 accumulators; B reused 4x.
__global__ __launch_bounds__(32)
void tconv_wmma_vec(const __bf16* __restrict__ Xt, const __bf16* __restrict__ Wp,
                    const float* __restrict__ bias, __bf16* __restrict__ Y,
                    int Cin, int Cout) {
  __shared__ __align__(16) __bf16 sm[64 * 16];
  const int bn = blockIdx.x, co0 = blockIdx.y << 6, t0 = blockIdx.z << 4;
  const int lane = threadIdx.x & 31, half = lane >> 4, mn = lane & 15;
  const __bf16* Xb = Xt + (size_t)bn * TLEN * Cin;

  v8f acc[4] = {};
  for (int kw = 0; kw < 3; ++kw) {
    const int tt = t0 + mn + kw - 1;
    const bool ok = (tt >= 0) && (tt < TLEN);
    const __bf16* xrow = Xb + (size_t)(ok ? tt : 0) * Cin;
    const __bf16* wkw = Wp + (size_t)kw * Cout * Cin;
    for (int kk = 0; kk < Cin; kk += 32) {
      v16bf b;
      if (ok) {
        frag_load8(b, 0, xrow + kk + 8 * half);
        frag_load8(b, 8, xrow + kk + 8 * half + 16);
      } else {
#pragma unroll
        for (int e = 0; e < 16; ++e) b[e] = (__bf16)0.0f;
      }
#pragma unroll
      for (int q = 0; q < 4; ++q) {
        const __bf16* wr = wkw + (size_t)(co0 + q * 16 + mn) * Cin + kk + 8 * half;
        v16bf a;
        frag_load8(a, 0, wr);
        frag_load8(a, 8, wr + 16);
        acc[q] = wmma_bf16(a, b, acc[q]);
      }
    }
  }
  // stage D tiles in LDS, then coalesced u16 row stores (T=174 => 2B alignment only)
#pragma unroll
  for (int q = 0; q < 4; ++q)
#pragma unroll
    for (int j = 0; j < 8; ++j) {
      int cr = q * 16 + j + 8 * half;
      sm[cr * 16 + mn] = (__bf16)(acc[q][j] + bias[co0 + cr]);
    }
  __syncthreads();
  __bf16* Yb = Y + (size_t)bn * Cout * TLEN;
#pragma unroll
  for (int i = 0; i < 32; ++i) {
    int idx = lane + 32 * i;              // 0..1023 over 64 rows x 16 cols
    int row = idx >> 4, col = idx & 15;
    int t = t0 + col;
    if (t < TLEN) Yb[(size_t)(co0 + row) * TLEN + t] = sm[row * 16 + col];
  }
}

// ---------------- tridiagonal Laplacian stencils (x8 vectorized) ----------------
__device__ __forceinline__ void l_coefs(int m, float& cd, float& cu, float& cl) {
  float dm = (m == 0 || m == NNODE - 1) ? 2.0f : 3.0f;
  float ism = rsqrtf(dm + 1e-8f);
  cd = 1.0f - ism * ism;
  cu = 0.0f; cl = 0.0f;
  if (m > 0) {
    float dn = (m - 1 == 0) ? 2.0f : 3.0f;
    cu = -ism * rsqrtf(dn + 1e-8f);
  }
  if (m < NNODE - 1) {
    float dn = (m + 1 == NNODE - 1) ? 2.0f : 3.0f;
    cl = -ism * rsqrtf(dn + 1e-8f);
  }
}

__global__ void cheb_t1_vec(const __bf16* __restrict__ T0, __bf16* __restrict__ T1,
                            int C, size_t nvec) {
  size_t v = (size_t)blockIdx.x * blockDim.x + threadIdx.x;
  if (v >= nvec) return;
  size_t idx = v * 8;
  size_t r = idx / (size_t)C;
  int m = (int)((r / TLEN) % NNODE);
  float cd, cu, cl; l_coefs(m, cd, cu, cl);
  const size_t ns = (size_t)TLEN * C;
  bf16x8 x0 = *(const bf16x8*)(T0 + idx);
  bf16x8 xu = *(const bf16x8*)(T0 + idx - (m > 0 ? ns : 0));          // cu==0 at m==0
  bf16x8 xl = *(const bf16x8*)(T0 + idx + (m < NNODE - 1 ? ns : 0));  // cl==0 at m==39
  bf16x8 res;
#pragma unroll
  for (int e = 0; e < 8; ++e)
    res[e] = (__bf16)(cd * (float)x0[e] + cu * (float)xu[e] + cl * (float)xl[e]);
  *(bf16x8*)(T1 + idx) = res;
}

__global__ void cheb_t2_vec(const __bf16* __restrict__ T0, const __bf16* __restrict__ T1,
                            __bf16* __restrict__ T2, int C, size_t nvec) {
  size_t v = (size_t)blockIdx.x * blockDim.x + threadIdx.x;
  if (v >= nvec) return;
  size_t idx = v * 8;
  size_t r = idx / (size_t)C;
  int m = (int)((r / TLEN) % NNODE);
  float cd, cu, cl; l_coefs(m, cd, cu, cl);
  const size_t ns = (size_t)TLEN * C;
  bf16x8 x1 = *(const bf16x8*)(T1 + idx);
  bf16x8 xu = *(const bf16x8*)(T1 + idx - (m > 0 ? ns : 0));
  bf16x8 xl = *(const bf16x8*)(T1 + idx + (m < NNODE - 1 ? ns : 0));
  bf16x8 x0 = *(const bf16x8*)(T0 + idx);
  bf16x8 res;
#pragma unroll
  for (int e = 0; e < 8; ++e)
    res[e] = (__bf16)(2.0f * (cd * (float)x1[e] + cu * (float)xu[e] + cl * (float)xl[e])
                      - (float)x0[e]);
  *(bf16x8*)(T2 + idx) = res;
}

// ---------------- Chebyshev channel mix: 16(r) x 64(co) tiles, A reused 4x ----------------
__global__ __launch_bounds__(32)
void mix_wmma_vec(const __bf16* __restrict__ T0s, const __bf16* __restrict__ T1s,
                  const __bf16* __restrict__ T2s, const __bf16* __restrict__ Wpk,
                  const float* __restrict__ bias, __bf16* __restrict__ Y, int C) {
  __shared__ __align__(16) __bf16 sm[16 * 64];
  const int r0 = blockIdx.x << 4, co0 = blockIdx.y << 6;
  const int lane = threadIdx.x & 31, half = lane >> 4, mn = lane & 15;
  const int K3 = 3 * C;

  v8f acc[4] = {};
  for (int kk = 0; kk < K3; kk += 32) {
    int s = kk / C, kb = kk - s * C;                  // never straddles (C % 32 == 0)
    const __bf16* S = (s == 0) ? T0s : ((s == 1) ? T1s : T2s);
    const __bf16* ar = S + (size_t)(r0 + mn) * C + kb + 8 * half;
    if (kk + 32 < K3) __builtin_prefetch(ar + 32, 0, 1);   // global_prefetch_b8
    v16bf a;
    frag_load8(a, 0, ar);
    frag_load8(a, 8, ar + 16);
    const int q32 = kk >> 5;
#pragma unroll
    for (int q = 0; q < 4; ++q) {
      const __bf16* br = Wpk + ((size_t)q32 * C + (co0 + q * 16 + mn)) * 32 + 8 * half;
      v16bf b;
      frag_load8(b, 0, br);
      frag_load8(b, 8, br + 16);
      acc[q] = wmma_bf16(a, b, acc[q]);
    }
  }
  // stage [16r x 64co] in LDS, then contiguous b128 stores (C % 64 == 0 => aligned)
#pragma unroll
  for (int q = 0; q < 4; ++q) {
    const float bs = bias[co0 + q * 16 + mn];
#pragma unroll
    for (int j = 0; j < 8; ++j)
      sm[(j + 8 * half) * 64 + q * 16 + mn] = (__bf16)(acc[q][j] + bs);
  }
  __syncthreads();
#pragma unroll
  for (int i = 0; i < 4; ++i) {
    int c8 = lane + 32 * i;                // 0..127 chunks of 8 bf16
    int row = c8 >> 3, seg = c8 & 7;
    bf16x8 val = *(const bf16x8*)(sm + row * 64 + seg * 8);
    *(bf16x8*)(Y + (size_t)(r0 + row) * C + co0 + seg * 8) = val;
  }
}

// ---------------- BatchNorm(inference) + ReLU (x8 vectorized) ----------------
__global__ void bn_relu_vec(const __bf16* __restrict__ Yin, __bf16* __restrict__ Yout,
                            const float* __restrict__ gamma, const float* __restrict__ beta,
                            const float* __restrict__ mean, const float* __restrict__ var,
                            int C, size_t nvec) {
  size_t v = (size_t)blockIdx.x * blockDim.x + threadIdx.x;
  if (v >= nvec) return;
  size_t idx = v * 8;
  int c0 = (int)(idx % (size_t)C);                   // C % 8 == 0 => c0..c0+7 valid
  bf16x8 y = *(const bf16x8*)(Yin + idx);
  bf16x8 res;
#pragma unroll
  for (int e = 0; e < 8; ++e) {
    int c = c0 + e;
    float t = ((float)y[e] - mean[c]) * (gamma[c] * rsqrtf(var[c] + 1e-5f)) + beta[c];
    res[e] = (__bf16)fmaxf(t, 0.0f);
  }
  *(bf16x8*)(Yout + idx) = res;
}

// ---------------- global mean pool over (N,T) ----------------
__global__ void pool_kernel(const __bf16* __restrict__ H, float* __restrict__ pooled) {
  int tid = blockIdx.x * blockDim.x + threadIdx.x;
  if (tid >= BATCH * 256) return;
  int b = tid >> 8, c = tid & 255;
  const int NT = NNODE * TLEN;
  const __bf16* Hb = H + (size_t)b * NT * 256 + c;
  float s = 0.0f;
  for (int j = 0; j < NT; ++j) s += (float)Hb[(size_t)j * 256];
  pooled[tid] = s / (float)NT;
}

// ---------------- FC ----------------
__global__ void fc_kernel(const float* __restrict__ pooled, const float* __restrict__ fw,
                          const float* __restrict__ fb, float* __restrict__ out) {
  int tid = blockIdx.x * blockDim.x + threadIdx.x;
  if (tid >= BATCH * 10) return;
  int b = tid / 10, j = tid - b * 10;
  float s = fb[j];
  for (int c = 0; c < 256; ++c) s += pooled[b * 256 + c] * fw[j * 256 + c];
  out[tid] = s;
}

// ---------------- driver ----------------
extern "C" void kernel_launch(void* const* d_in, const int* in_sizes, int n_in,
                              void* d_out, int out_size, void* d_ws, size_t ws_size,
                              hipStream_t stream) {
  (void)in_sizes; (void)n_in; (void)out_size; (void)ws_size;
  const float* x = (const float*)d_in[0];
  auto P = [&](int blkI, int j) { return (const float*)d_in[1 + 10 * blkI + j]; };
  const float* fc_w = (const float*)d_in[31];
  const float* fc_b = (const float*)d_in[32];

  const size_t cap = (size_t)RROWS * 256;
  __bf16* A   = (__bf16*)d_ws;         // activation ping
  __bf16* Bb  = A + cap;               // activation pong
  __bf16* T1s = Bb + cap;              // cheb T1 slab / tconv2 transpose scratch
  __bf16* T2s = T1s + cap;             // cheb T2 slab / tconv1 transpose scratch
  __bf16* W1  = T2s + cap;
  __bf16* WC  = W1 + 200000;
  __bf16* W2  = WC + 200000;
  float*  pooled = (float*)(W2 + 200000);

  // input [32,1,40,174]: transpose(0,2,3,1) is identity reshape since C==1
  cast_f32_bf16<<<(RROWS + 255) / 256, 256, 0, stream>>>(x, A, RROWS);

  const int Cs[4] = {1, 64, 128, 256};
  const unsigned tz = (TLEN + 15) / 16;   // 11
  for (int i = 0; i < 3; ++i) {
    const int Cin = Cs[i], Cout = Cs[i + 1];
    const size_t total = (size_t)RROWS * Cout;
    const unsigned vb = (unsigned)((total / 8 + 255) / 256);

    // ---- temporal conv 1 ----
    if (Cin < 32) {  // block 1: scalar im2col path
      int n = Cout * Cin * 3;
      cast_f32_bf16<<<(n + 255) / 256, 256, 0, stream>>>(P(i, 0), W1, n);
      tconv_wmma_scalar<<<dim3(BN, Cout / 16, tz), 32, 0, stream>>>(
          A, W1, P(i, 1), Bb, Cin, Cout);
    } else {
      int n = 3 * Cout * Cin;
      pack_tconv_w<<<(n + 255) / 256, 256, 0, stream>>>(P(i, 0), W1, Cin, Cout);
      transpose_ct<<<dim3(BN, Cin / 16, tz), 256, 0, stream>>>(A, T2s, Cin);
      tconv_wmma_vec<<<dim3(BN, Cout / 64, tz), 32, 0, stream>>>(
          T2s, W1, P(i, 1), Bb, Cin, Cout);
    }

    // ---- Chebyshev terms (tridiagonal L stencils) ----
    cheb_t1_vec<<<vb, 256, 0, stream>>>(Bb, T1s, Cout, total / 8);
    cheb_t2_vec<<<vb, 256, 0, stream>>>(Bb, T1s, T2s, Cout, total / 8);

    // ---- channel mix GEMM ----
    { int n = 3 * Cout * Cout;
      pack_mix_w<<<(n + 255) / 256, 256, 0, stream>>>(P(i, 2), WC, Cout); }
    mix_wmma_vec<<<dim3(RROWS / 16, Cout / 64), 32, 0, stream>>>(
        Bb, T1s, T2s, WC, P(i, 3), A, Cout);

    // ---- temporal conv 2 (Cin = Cout >= 64) ----
    { int n = 3 * Cout * Cout;
      pack_tconv_w<<<(n + 255) / 256, 256, 0, stream>>>(P(i, 4), W2, Cout, Cout); }
    transpose_ct<<<dim3(BN, Cout / 16, tz), 256, 0, stream>>>(A, T1s, Cout);
    tconv_wmma_vec<<<dim3(BN, Cout / 64, tz), 32, 0, stream>>>(
        T1s, W2, P(i, 5), Bb, Cout, Cout);

    // ---- BN + ReLU ----
    bn_relu_vec<<<vb, 256, 0, stream>>>(Bb, A, P(i, 6), P(i, 7), P(i, 8), P(i, 9),
                                        Cout, total / 8);
  }

  pool_kernel<<<(BATCH * 256 + 255) / 256, 256, 0, stream>>>(A, pooled);
  fc_kernel<<<2, 256, 0, stream>>>(pooled, fc_w, fc_b, (float*)d_out);
}
